// GroupedQueryAttention_19387482374694
// MI455X (gfx1250) — compile-verified, matrix operand order checked by reading, not run on hardware
//
#include <hip/hip_runtime.h>

typedef __attribute__((ext_vector_type(16))) __bf16 bf16x16;
typedef __attribute__((ext_vector_type(8)))  float  f32x8;
typedef unsigned short u16;
typedef unsigned int   u32;
typedef unsigned long long u64;

constexpr int cB   = 2;
constexpr int cT   = 2048;
constexpr int cE   = 4096;
constexpr int cHQ  = 32;
constexpr int cD   = 128;
constexpr int cBT  = cB * cT;   // 4096
constexpr int cKV  = cE / 4;    // 1024 (KV projection width)

// ---------- helpers ----------

__device__ __forceinline__ u16 f2bf(float f) {          // RNE f32 -> bf16
  u32 u = __float_as_uint(f);
  u32 r = u + 0x7FFFu + ((u >> 16) & 1u);
  return (u16)(r >> 16);
}

union FragU { bf16x16 v; uint4 q[2]; };

// Load a 16x32 bf16 WMMA operand fragment from row-major memory.
// ISA layout (16-bit A/B): lanes 0-15 -> row, K {k0..k0+7} and {k0+16..k0+23};
// lanes 16-31 -> same rows, K {k0+8..k0+15} and {k0+24..k0+31}.
__device__ __forceinline__ bf16x16 load_frag_g(const u16* __restrict__ base,
                                               long row0, long ld, long k0) {
  int lane = threadIdx.x & 31;
  long r  = row0 + (lane & 15);
  long kc = k0 + (long)((lane >> 4) << 3);
  const uint4* p = (const uint4*)(base + r * ld + kc);
  FragU f;
  f.q[0] = p[0];   // K = kc .. kc+7
  f.q[1] = p[2];   // K = kc+16 .. kc+23
  return f.v;
}

__device__ __forceinline__ bf16x16 load_frag_lds(const u16* base, int ld, int k0) {
  int lane = threadIdx.x & 31;
  int r  = lane & 15;
  int kc = k0 + ((lane >> 4) << 3);
  const uint4* p = (const uint4*)(base + r * ld + kc);
  FragU f;
  f.q[0] = p[0];
  f.q[1] = p[2];
  return f.v;
}

__device__ __forceinline__ f32x8 wmma_bf16(bf16x16 a, bf16x16 b, f32x8 c) {
  return __builtin_amdgcn_wmma_f32_16x16x32_bf16(false, a, false, b,
                                                 (short)0, c, false, false);
}

// Async global -> LDS copy of 16 bytes per lane (CDNA5 path, ASYNCcnt-tracked).
__device__ __forceinline__ void async_copy_b128(unsigned lds_off, const void* g) {
  asm volatile("global_load_async_to_lds_b128 %0, %1, off"
               :: "v"(lds_off), "v"((u64)(size_t)g) : "memory");
}

// ---------- f32 -> bf16 cast (8 elems / thread) ----------

__global__ void cast_f32_to_bf16(const float* __restrict__ in,
                                 u16* __restrict__ out, int n) {
  int i = (blockIdx.x * blockDim.x + threadIdx.x) * 8;
  if (i >= n) return;
  float4 f0 = *(const float4*)(in + i);
  float4 f1 = *(const float4*)(in + i + 4);
  union { u16 s[8]; uint4 q; } o;
  o.s[0] = f2bf(f0.x); o.s[1] = f2bf(f0.y); o.s[2] = f2bf(f0.z); o.s[3] = f2bf(f0.w);
  o.s[4] = f2bf(f1.x); o.s[5] = f2bf(f1.y); o.s[6] = f2bf(f1.z); o.s[7] = f2bf(f1.w);
  *(uint4*)(out + i) = o.q;
}

// ---------- GEMM: C[M,N] = A[M,K] * W[N,K]^T  (all bf16, f32 acc) ----------
// Block: 256 thr = 8 waves in 2x4; wave tile 64x64 (4x4 WMMA tiles).
// Block tile 128x256, K-step 32, double-buffered LDS staging filled with
// global_load_async_to_lds_b128 (6 async instr / wave / stage).
// MODE 0: bf16 row-major C. MODE 1: bf16 transposed C[n][m] (packed b128
// stores). MODE 2: f32 C + bias.

constexpr int LDP = 40;   // LDS row pitch in u16: 80B rows -> 16B aligned,
                          // conflict-free for 16-lane row-parallel b128 reads

template <int MODE>
__global__ __launch_bounds__(256)
void gemm_bf16_wmma(const u16* __restrict__ A, const u16* __restrict__ W,
                    void* __restrict__ Cout, const float* __restrict__ bias,
                    int M, int N, int K, int ldct) {
  (void)M;
  __shared__ __align__(16) u16 sA[2][128][LDP];   // 20 KB
  __shared__ __align__(16) u16 sB[2][256][LDP];   // 40 KB
  const int tid  = threadIdx.x;
  const int wave = tid >> 5;
  const int lane = tid & 31;
  const int wm = wave >> 2, wn = wave & 3;
  const long mblk = (long)blockIdx.y * 128;
  const long nblk = (long)blockIdx.x * 256;
  const int m0 = wm * 64, n0 = wn * 64;

  // Stage one 128x32 A tile + 256x32 B tile into LDS buffer `buf`.
  // A: 512 16B-chunks -> 2 per thread; B: 1024 chunks -> 4 per thread.
  auto stage = [&](int buf, int k0) {
    for (int c = tid; c < 512; c += 256) {
      int row = c >> 2, col = (c & 3) * 8;
      async_copy_b128((unsigned)(size_t)&sA[buf][row][col],
                      A + (mblk + row) * (long)K + k0 + col);
    }
    for (int c = tid; c < 1024; c += 256) {
      int row = c >> 2, col = (c & 3) * 8;
      async_copy_b128((unsigned)(size_t)&sB[buf][row][col],
                      W + (nblk + row) * (long)K + k0 + col);
    }
  };

  f32x8 acc[4][4];
#pragma unroll
  for (int i = 0; i < 4; ++i)
#pragma unroll
    for (int j = 0; j < 4; ++j)
#pragma unroll
      for (int e = 0; e < 8; ++e) acc[i][j][e] = 0.0f;

  stage(0, 0);
  const int nk = K >> 5;
  for (int kk = 0; kk < nk; ++kk) {
    const int buf = kk & 1;
    if (kk + 1 < nk) {
      stage(buf ^ 1, (kk + 1) << 5);
      // async loads complete in order: <=6 outstanding means current buffer
      // (the previous 6 per-wave async ops) has landed in LDS.
      asm volatile("s_wait_asynccnt 0x6" ::: "memory");
    } else {
      asm volatile("s_wait_asynccnt 0x0" ::: "memory");
    }
    __syncthreads();          // every wave's share of `buf` is in LDS

    bf16x16 a[4], b[4];
#pragma unroll
    for (int i = 0; i < 4; ++i)
      a[i] = load_frag_lds(&sA[buf][m0 + i * 16][0], LDP, 0);
#pragma unroll
    for (int j = 0; j < 4; ++j)
      b[j] = load_frag_lds(&sB[buf][n0 + j * 16][0], LDP, 0);
#pragma unroll
    for (int i = 0; i < 4; ++i)
#pragma unroll
      for (int j = 0; j < 4; ++j)
        acc[i][j] = wmma_bf16(a[i], b[j], acc[i][j]);

    __syncthreads();          // all reads of `buf` done before it is restaged
  }

  const int cn = lane & 15, hi = lane >> 4;   // D-layout: lane -> n, vgpr -> m
  if (MODE == 0) {
    u16* C = (u16*)Cout;
#pragma unroll
    for (int i = 0; i < 4; ++i)
#pragma unroll
      for (int j = 0; j < 4; ++j) {
        long n = nblk + n0 + j * 16 + cn;
#pragma unroll
        for (int v = 0; v < 8; ++v) {
          long m = mblk + m0 + i * 16 + v + 8 * hi;
          C[m * (long)N + n] = f2bf(acc[i][j][v]);
        }
      }
  } else if (MODE == 1) {
    u16* C = (u16*)Cout;
#pragma unroll
    for (int i = 0; i < 4; ++i)
#pragma unroll
      for (int j = 0; j < 4; ++j) {
        long n  = nblk + n0 + j * 16 + cn;
        long mb = mblk + m0 + i * 16 + 8 * hi;   // 8 consecutive m per lane
        union { u16 s[8]; uint4 q; } t;
#pragma unroll
        for (int v = 0; v < 8; ++v) t.s[v] = f2bf(acc[i][j][v]);
        *(uint4*)(C + n * (long)ldct + mb) = t.q;   // packed b128 store
      }
  } else {
    float* C = (float*)Cout;
#pragma unroll
    for (int j = 0; j < 4; ++j) {
      long n = nblk + n0 + j * 16 + cn;
      float bv = bias[n];
#pragma unroll
      for (int i = 0; i < 4; ++i)
#pragma unroll
        for (int v = 0; v < 8; ++v) {
          long m = mblk + m0 + i * 16 + v + 8 * hi;
          C[m * (long)N + n] = acc[i][j][v] + bv;
        }
    }
  }
}

// ---------- Flash GQA attention ----------
// grid.y = b*32 + hq ; grid.x = q-tile of 128 ; block = 4 waves * 32 queries.
// Q: [bt][E] bf16 (head slice hq*128). K: [bt][1024]. Vt: [chan][bt] (transposed).
// Y: [bt][E] bf16 at head slice hq*128. K/V are L2-resident (16 MB total bf16).

__global__ __launch_bounds__(128)
void gqa_flash_attn(const u16* __restrict__ Q, const u16* __restrict__ Kc,
                    const u16* __restrict__ Vt, u16* __restrict__ Y) {
  const int wave = threadIdx.x >> 5;
  const int lane = threadIdx.x & 31;
  const int cn = lane & 15, hi = lane >> 4;
  const int bh = blockIdx.y;
  const int b  = bh >> 5;
  const int hq = bh & 31;
  const int hk = hq >> 2;                   // 4 Q-heads per KV-head
  const long qrow0 = (long)b * cT + (long)blockIdx.x * 128 + (long)wave * 32;

  const u16* Qb = Q  + (long)hq * cD;                         // ld = cE
  const u16* Kb = Kc + (long)hk * cD;                         // ld = cKV
  const u16* Vb = Vt + ((long)hk * cD) * cBT + (long)b * cT;  // ld = cBT

  bf16x16 qf[2][4];                          // 32 queries x 128 d, resident
#pragma unroll
  for (int i = 0; i < 2; ++i)
#pragma unroll
    for (int kf = 0; kf < 4; ++kf)
      qf[i][kf] = load_frag_g(Qb, qrow0 + i * 16, cE, kf * 32);

  f32x8 o[2][8];
  float rmax[2][8], rsum[2][8];
#pragma unroll
  for (int i = 0; i < 2; ++i)
#pragma unroll
    for (int v = 0; v < 8; ++v) { rmax[i][v] = -1e30f; rsum[i][v] = 0.0f; }
#pragma unroll
  for (int i = 0; i < 2; ++i)
#pragma unroll
    for (int j = 0; j < 8; ++j)
#pragma unroll
      for (int e = 0; e < 8; ++e) o[i][j][e] = 0.0f;

  __shared__ __align__(16) u16 sP[4][32][72];   // per-wave P transpose staging

  const float sc  = 0.08838834764831845f;    // 1/sqrt(128)
  const float L2E = 1.4426950408889634f;

  for (int kb0 = 0; kb0 < cT; kb0 += 64) {
    // S = Q * K^T for a 32q x 64k chunk
    f32x8 s[2][4];
#pragma unroll
    for (int i = 0; i < 2; ++i)
#pragma unroll
      for (int j = 0; j < 4; ++j)
#pragma unroll
        for (int e = 0; e < 8; ++e) s[i][j][e] = 0.0f;

#pragma unroll
    for (int j = 0; j < 4; ++j)
#pragma unroll
      for (int kf = 0; kf < 4; ++kf) {
        bf16x16 kfr = load_frag_g(Kb, (long)b * cT + kb0 + j * 16, cKV, kf * 32);
#pragma unroll
        for (int i = 0; i < 2; ++i)
          s[i][j] = wmma_bf16(qf[i][kf], kfr, s[i][j]);
      }

    // streaming softmax; row = query = (v + 8*hi) within subtile i
#pragma unroll
    for (int i = 0; i < 2; ++i) {
      float p[4][8];
#pragma unroll
      for (int v = 0; v < 8; ++v) {
        float mx = fmaxf(fmaxf(s[i][0][v], s[i][1][v]),
                         fmaxf(s[i][2][v], s[i][3][v]));
#pragma unroll
        for (int off = 1; off < 16; off <<= 1)
          mx = fmaxf(mx, __shfl_xor(mx, off, 32));
        mx *= sc;
        float nm = fmaxf(rmax[i][v], mx);
        float alpha = exp2f((rmax[i][v] - nm) * L2E);
        float ls = 0.0f;
#pragma unroll
        for (int j = 0; j < 4; ++j) {
          float pv = exp2f((s[i][j][v] * sc - nm) * L2E);
          p[j][v] = pv;
          ls += pv;
        }
#pragma unroll
        for (int off = 1; off < 16; off <<= 1)
          ls += __shfl_xor(ls, off, 32);
        rsum[i][v] = rsum[i][v] * alpha + ls;
        rmax[i][v] = nm;
#pragma unroll
        for (int j2 = 0; j2 < 8; ++j2) o[i][j2][v] *= alpha;
      }
      // C-layout (lane=key) -> LDS, to be reloaded in A-layout (lane=query)
#pragma unroll
      for (int j = 0; j < 4; ++j)
#pragma unroll
        for (int v = 0; v < 8; ++v)
          sP[wave][i * 16 + v + 8 * hi][j * 16 + cn] = f2bf(p[j][v]);
    }
    __syncthreads();

    bf16x16 pf[2][2];
#pragma unroll
    for (int i = 0; i < 2; ++i)
#pragma unroll
      for (int kc = 0; kc < 2; ++kc)
        pf[i][kc] = load_frag_lds(&sP[wave][i * 16][0], 72, kc * 32);

    // O += P * V ; V^T layout makes B-operand rows contiguous
#pragma unroll
    for (int j2 = 0; j2 < 8; ++j2)
#pragma unroll
      for (int kc = 0; kc < 2; ++kc) {
        bf16x16 vf = load_frag_g(Vb, j2 * 16, cBT, kb0 + kc * 32);
#pragma unroll
        for (int i = 0; i < 2; ++i)
          o[i][j2] = wmma_bf16(pf[i][kc], vf, o[i][j2]);
      }
    __syncthreads();
  }

  // normalize and store y[bt][hq*128 + d]
#pragma unroll
  for (int i = 0; i < 2; ++i)
#pragma unroll
    for (int v = 0; v < 8; ++v) {
      float inv = 1.0f / rsum[i][v];
      long t = qrow0 + i * 16 + v + 8 * hi;
#pragma unroll
      for (int j2 = 0; j2 < 8; ++j2)
        Y[t * (long)cE + (long)hq * cD + j2 * 16 + cn] = f2bf(o[i][j2][v] * inv);
    }
}

// ---------- launch ----------

extern "C" void kernel_launch(void* const* d_in, const int* in_sizes, int n_in,
                              void* d_out, int out_size, void* d_ws, size_t ws_size,
                              hipStream_t stream) {
  (void)in_sizes; (void)n_in; (void)out_size; (void)ws_size;
  const float* x  = (const float*)d_in[0];
  const float* Wq = (const float*)d_in[1];
  const float* Wk = (const float*)d_in[2];
  const float* Wv = (const float*)d_in[3];
  const float* Wo = (const float*)d_in[4];
  const float* bo = (const float*)d_in[5];
  float* out = (float*)d_out;

  char* ws = (char*)d_ws;
  const size_t SZ_XB = (size_t)cBT * cE * 2;   // 32 MiB (x / y alias)
  const size_t SZ_WQ = (size_t)cE * cE * 2;    // 32 MiB (Wq / Wo alias)
  const size_t SZ_WK = (size_t)cKV * cE * 2;   // 8 MiB
  u16* xb  = (u16*)(ws);
  u16* Wqb = (u16*)(ws + SZ_XB);                        // later reused for Wo
  u16* Wkb = (u16*)(ws + SZ_XB + SZ_WQ);
  u16* Wvb = (u16*)(ws + SZ_XB + SZ_WQ + SZ_WK);
  u16* qb  = (u16*)(ws + SZ_XB + SZ_WQ + 2 * SZ_WK);
  u16* kb  = (u16*)(ws + 2 * SZ_XB + SZ_WQ + 2 * SZ_WK);
  u16* vtb = (u16*)(ws + 2 * SZ_XB + SZ_WQ + 3 * SZ_WK);
  u16* yb  = xb;                                        // x dead after projections
  // total workspace: 2*32 + 32 + 4*8 = 128 MiB

  const int nX  = cBT * cE;
  const int nWq = cE * cE;
  const int nWk = cKV * cE;
  cast_f32_to_bf16<<<dim3(nX  / 2048), dim3(256), 0, stream>>>(x,  xb,  nX);
  cast_f32_to_bf16<<<dim3(nWq / 2048), dim3(256), 0, stream>>>(Wq, Wqb, nWq);
  cast_f32_to_bf16<<<dim3(nWk / 2048), dim3(256), 0, stream>>>(Wk, Wkb, nWk);
  cast_f32_to_bf16<<<dim3(nWk / 2048), dim3(256), 0, stream>>>(Wv, Wvb, nWk);

  // q = x Wq^T  (4096x4096x4096)
  gemm_bf16_wmma<0><<<dim3(cE / 256,  cBT / 128), dim3(256), 0, stream>>>(
      xb, Wqb, qb, nullptr, cBT, cE, cE, 0);
  // k = x Wk^T  (4096x1024x4096)
  gemm_bf16_wmma<0><<<dim3(cKV / 256, cBT / 128), dim3(256), 0, stream>>>(
      xb, Wkb, kb, nullptr, cBT, cKV, cE, 0);
  // v^T = (x Wv^T)^T  stored as [chan][token]
  gemm_bf16_wmma<1><<<dim3(cKV / 256, cBT / 128), dim3(256), 0, stream>>>(
      xb, Wvb, vtb, nullptr, cBT, cKV, cE, cBT);

  // Wo cast into Wq's slot (Wq no longer needed)
  cast_f32_to_bf16<<<dim3(nWq / 2048), dim3(256), 0, stream>>>(Wo, Wqb, nWq);

  // attention: y written over x's slot (x no longer needed)
  gqa_flash_attn<<<dim3(cT / 128, cB * cHQ), dim3(128), 0, stream>>>(qb, kb, vtb, yb);

  // out = y Wo^T + bo (f32)
  gemm_bf16_wmma<2><<<dim3(cE / 256, cBT / 128), dim3(256), 0, stream>>>(
      yb, Wqb, out, bo, cBT, cE, cE, 0);
}
